// LinearAttnBlock_47107201302754
// MI455X (gfx1250) — matrix operand order repaired and checked by reading, not compiled, not run on hardware
//
#include <hip/hip_runtime.h>

// ---------------------------------------------------------------------------
// Types for CDNA5 WMMA (wave32): v_wmma_f32_16x16x32_bf16
// ---------------------------------------------------------------------------
typedef __bf16 bf16;
typedef __bf16 v16bf __attribute__((ext_vector_type(16)));
typedef __bf16 v8bf  __attribute__((ext_vector_type(8)));
typedef float  v8f   __attribute__((ext_vector_type(8)));
typedef float  v4f   __attribute__((ext_vector_type(4)));
typedef __bf16 v4bf  __attribute__((ext_vector_type(4)));

union BF16x16 { v16bf v; v8bf h[2]; };
union BF16x8  { v8bf v; bf16 e[8]; };
union BF16x4  { v4bf v; bf16 e[4]; };

#define C_DIM 256
#define N_DIM 4096   // 64*64
#define B_DIM 8

// ---------------------------------------------------------------------------
// Wave-level 64x64 GEMM tile: D(64x64,f32) += A(64xK) * B(Kx64)
//   A  : row-major [M x K]  (lda in elements)  -- row contiguous over K
//   Bt : row-major [N x K]  (ldb in elements)  -- i.e. B column-major
// Fragments use contiguous k order (k = 16*half + s) for both A and B.
// Inner loop keeps only ONE B fragment live at a time to limit VGPR pressure:
// peak ~ 128 (acc) + 32 (A frags) + 8 (B frag) + addressing.
// ---------------------------------------------------------------------------
__device__ __forceinline__ void wave_gemm_64x64(
    const bf16* __restrict__ A, int lda,
    const bf16* __restrict__ Bt, int ldb,
    int K, int m0, int n0, v8f acc[4][4])
{
    const int lane = threadIdx.x & 31;
    const int lm   = lane & 15;
    const int half = lane >> 4;

    const bf16* a0 = A  + (size_t)(m0 + lm) * lda + half * 16;
    const bf16* b0 = Bt + (size_t)(n0 + lm) * ldb + half * 16;

    for (int k = 0; k < K; k += 32) {
        BF16x16 af[4];
#pragma unroll
        for (int mi = 0; mi < 4; ++mi) {
            const bf16* ap = a0 + (size_t)(mi * 16) * lda + k;
            af[mi].h[0] = *(const v8bf*)(ap);
            af[mi].h[1] = *(const v8bf*)(ap + 8);
        }
#pragma unroll
        for (int ni = 0; ni < 4; ++ni) {
            v16bf bv = *(const v16bf*)(b0 + (size_t)(ni * 16) * ldb + k);
#pragma unroll
            for (int mi = 0; mi < 4; ++mi)
                acc[mi][ni] = __builtin_amdgcn_wmma_f32_16x16x32_bf16(
                    false, af[mi].v, false, bv,
                    (short)0, acc[mi][ni], false, false);
        }
    }
}

// ---------------------------------------------------------------------------
// f32 -> bf16 elementwise convert (vectorized x4)
// ---------------------------------------------------------------------------
__global__ void k_f32_to_bf16(const float* __restrict__ src,
                              bf16* __restrict__ dst, int n4)
{
    int i = blockIdx.x * 256 + threadIdx.x;   // index in float4 units
    if (i < n4) {
        v4f f = *(const v4f*)(src + (size_t)i * 4);
        BF16x4 u;
#pragma unroll
        for (int j = 0; j < 4; ++j) u.e[j] = (bf16)f[j];
        *(v4bf*)(dst + (size_t)i * 4) = u.v;
    }
}

// ---------------------------------------------------------------------------
// GroupNorm (32 groups of 8 channels over H*W) -> hT [B, N, C] bf16
// One workgroup per (group, batch); 256 threads.
// Each thread writes 16B (8 contiguous bf16 channels of its group) per n.
// ---------------------------------------------------------------------------
__global__ void k_groupnorm(const float* __restrict__ x,
                            const float* __restrict__ gamma,
                            const float* __restrict__ beta,
                            bf16* __restrict__ hT)
{
    const int g = blockIdx.x;      // 0..31
    const int b = blockIdx.y;      // 0..7
    const int tid = threadIdx.x;   // 0..255
    const int CG = 8;

    const float* xg = x + ((size_t)b * C_DIM + (size_t)g * CG) * N_DIM;

    float s = 0.f, ss = 0.f;
    for (int j = 0; j < CG; ++j) {
        const float* xc = xg + (size_t)j * N_DIM;
        for (int n = tid; n < N_DIM; n += 256) {
            float v = xc[n];
            s += v; ss += v * v;
        }
    }

    __shared__ float sh1[256], sh2[256];
    sh1[tid] = s; sh2[tid] = ss;
    __syncthreads();
    for (int off = 128; off > 0; off >>= 1) {
        if (tid < off) { sh1[tid] += sh1[tid + off]; sh2[tid] += sh2[tid + off]; }
        __syncthreads();
    }
    const float inv = 1.0f / (float)(CG * N_DIM);
    const float mean = sh1[0] * inv;
    const float var  = sh2[0] * inv - mean * mean;
    const float rs   = rsqrtf(var + 1e-6f);

    float ga[8], be[8];
#pragma unroll
    for (int j = 0; j < 8; ++j) {
        float gmm = gamma[g * CG + j];
        ga[j] = gmm * rs;
        be[j] = beta[g * CG + j] - mean * rs * gmm;
    }

    bf16* hb = hT + (size_t)b * N_DIM * C_DIM + (size_t)g * CG;
    for (int n = tid; n < N_DIM; n += 256) {
        BF16x8 u;
#pragma unroll
        for (int j = 0; j < 8; ++j) {
            float v = xg[(size_t)j * N_DIM + n];
            u.e[j] = (bf16)(v * ga[j] + be[j]);
        }
        *(v8bf*)(hb + (size_t)n * C_DIM) = u.v;
    }
}

// ---------------------------------------------------------------------------
// Q/K/V projection GEMM: out[c,n] = W[c,:] . h[:,n] + bias[c]  (per batch)
//   A  = Wb   [C x C] bf16 row-major
//   Bt = hT   [N x C] bf16 (per-batch)
// do_elu: apply elu(x)+1  (x>0 ? x+1 : exp(x))
// do_transpose: store out as [N x C] (for Q), else [C x N]
// grid (16, 4, B) x 128 threads (4 waves; wave -> n sub-tile)
// ---------------------------------------------------------------------------
__global__ void __launch_bounds__(128, 1)
k_qkv_gemm(const bf16* __restrict__ Wb,
           const float* __restrict__ bias,
           const bf16* __restrict__ hT,
           bf16* __restrict__ out,
           int do_elu, int do_transpose)
{
    const int b    = blockIdx.z;
    const int wave = threadIdx.x >> 5;
    const int n0   = (blockIdx.x * 4 + wave) * 64;
    const int m0   = blockIdx.y * 64;

    const bf16* Bt = hT + (size_t)b * N_DIM * C_DIM;

    v8f acc[4][4] = {};
    wave_gemm_64x64(Wb, C_DIM, Bt, C_DIM, C_DIM, m0, n0, acc);

    const int lane = threadIdx.x & 31;
    const int lm = lane & 15, half = lane >> 4;

#pragma unroll
    for (int mi = 0; mi < 4; ++mi)
#pragma unroll
        for (int ni = 0; ni < 4; ++ni)
#pragma unroll
            for (int r = 0; r < 8; ++r) {
                int row = m0 + mi * 16 + half * 8 + r;
                int col = n0 + ni * 16 + lm;
                float v = acc[mi][ni][r] + bias[row];
                if (do_elu) v = (v > 0.f) ? (v + 1.f) : __expf(v);
                bf16 o = (bf16)v;
                if (do_transpose)
                    out[(size_t)b * N_DIM * C_DIM + (size_t)col * C_DIM + row] = o;
                else
                    out[(size_t)b * C_DIM * N_DIM + (size_t)row * N_DIM + col] = o;
            }
}

// ---------------------------------------------------------------------------
// kv[c,d] = sum_n K[c,n] * V[d,n]   (per batch, K-dim = 4096)
//   A  = Kb [C x N] row-major, Bt = Vb [C x N] row-major
// grid (1, 4, B) x 128 threads
// ---------------------------------------------------------------------------
__global__ void __launch_bounds__(128, 1)
k_kv_gemm(const bf16* __restrict__ Kb,
          const bf16* __restrict__ Vb,
          float* __restrict__ KV)
{
    const int b    = blockIdx.z;
    const int wave = threadIdx.x >> 5;
    const int n0   = wave * 64;
    const int m0   = blockIdx.y * 64;

    const bf16* A  = Kb + (size_t)b * C_DIM * N_DIM;
    const bf16* Bt = Vb + (size_t)b * C_DIM * N_DIM;

    v8f acc[4][4] = {};
    wave_gemm_64x64(A, N_DIM, Bt, N_DIM, N_DIM, m0, n0, acc);

    const int lane = threadIdx.x & 31;
    const int lm = lane & 15, half = lane >> 4;
    float* o = KV + (size_t)b * C_DIM * C_DIM;
#pragma unroll
    for (int mi = 0; mi < 4; ++mi)
#pragma unroll
        for (int ni = 0; ni < 4; ++ni)
#pragma unroll
            for (int r = 0; r < 8; ++r) {
                int row = m0 + mi * 16 + half * 8 + r;
                int col = n0 + ni * 16 + lm;
                o[(size_t)row * C_DIM + col] = acc[mi][ni][r];
            }
}

// ---------------------------------------------------------------------------
// M[o,c] = sum_d wo[o,d] * kv[c,d]   (wo folded into kv; 256^3 per batch)
//   A = Wob [C x C] bf16 row-major (shared), Bt = KVb [C x C] bf16 row-major
// grid (1, 4, B) x 128 threads; output bf16 row-major (A of final GEMM)
// ---------------------------------------------------------------------------
__global__ void __launch_bounds__(128, 1)
k_m_gemm(const bf16* __restrict__ Wob,
         const bf16* __restrict__ KVb,
         bf16* __restrict__ Mb)
{
    const int b    = blockIdx.z;
    const int wave = threadIdx.x >> 5;
    const int n0   = wave * 64;
    const int m0   = blockIdx.y * 64;

    const bf16* Bt = KVb + (size_t)b * C_DIM * C_DIM;

    v8f acc[4][4] = {};
    wave_gemm_64x64(Wob, C_DIM, Bt, C_DIM, C_DIM, m0, n0, acc);

    const int lane = threadIdx.x & 31;
    const int lm = lane & 15, half = lane >> 4;
    bf16* o = Mb + (size_t)b * C_DIM * C_DIM;
#pragma unroll
    for (int mi = 0; mi < 4; ++mi)
#pragma unroll
        for (int ni = 0; ni < 4; ++ni)
#pragma unroll
            for (int r = 0; r < 8; ++r) {
                int row = m0 + mi * 16 + half * 8 + r;
                int col = n0 + ni * 16 + lm;
                o[(size_t)row * C_DIM + col] = (bf16)acc[mi][ni][r];
            }
}

// ---------------------------------------------------------------------------
// out[o,n] = x[o,n] + sum_c M[o,c]*q[c,n] + bo[o]
//   A = Mb [C x C] bf16 (per batch), Bt = QT [N x C] bf16 (per batch)
// grid (16, 4, B) x 128 threads; output f32
// ---------------------------------------------------------------------------
__global__ void __launch_bounds__(128, 1)
k_out_gemm(const bf16* __restrict__ Mb,
           const bf16* __restrict__ QT,
           const float* __restrict__ x,
           const float* __restrict__ bo,
           float* __restrict__ out)
{
    const int b    = blockIdx.z;
    const int wave = threadIdx.x >> 5;
    const int n0   = (blockIdx.x * 4 + wave) * 64;
    const int m0   = blockIdx.y * 64;

    const bf16* A  = Mb + (size_t)b * C_DIM * C_DIM;
    const bf16* Bt = QT + (size_t)b * N_DIM * C_DIM;

    v8f acc[4][4] = {};
    wave_gemm_64x64(A, C_DIM, Bt, C_DIM, C_DIM, m0, n0, acc);

    const int lane = threadIdx.x & 31;
    const int lm = lane & 15, half = lane >> 4;
    const float* xb = x   + (size_t)b * C_DIM * N_DIM;
    float*       ob = out + (size_t)b * C_DIM * N_DIM;
#pragma unroll
    for (int mi = 0; mi < 4; ++mi)
#pragma unroll
        for (int ni = 0; ni < 4; ++ni)
#pragma unroll
            for (int r = 0; r < 8; ++r) {
                int row = m0 + mi * 16 + half * 8 + r;
                int col = n0 + ni * 16 + lm;
                size_t idx = (size_t)row * N_DIM + col;
                ob[idx] = acc[mi][ni][r] + xb[idx] + bo[row];
            }
}

// ---------------------------------------------------------------------------
// Launch: GN -> QKV -> KV -> M=wo*kv^T -> out = x + M*Q + bo
// ---------------------------------------------------------------------------
extern "C" void kernel_launch(void* const* d_in, const int* in_sizes, int n_in,
                              void* d_out, int out_size, void* d_ws, size_t ws_size,
                              hipStream_t stream)
{
    const float* x     = (const float*)d_in[0];
    const float* gamma = (const float*)d_in[1];
    const float* beta  = (const float*)d_in[2];
    const float* wq    = (const float*)d_in[3];
    const float* bq    = (const float*)d_in[4];
    const float* wk    = (const float*)d_in[5];
    const float* bk    = (const float*)d_in[6];
    const float* wv    = (const float*)d_in[7];
    const float* bv    = (const float*)d_in[8];
    const float* wo    = (const float*)d_in[9];
    const float* bo    = (const float*)d_in[10];
    float* out = (float*)d_out;

    char* ws = (char*)d_ws;
    const size_t BNC = (size_t)B_DIM * N_DIM * C_DIM;   // 8M elements
    const size_t BCC = (size_t)B_DIM * C_DIM * C_DIM;   // 512K elements
    const size_t CC  = (size_t)C_DIM * C_DIM;           // 64K elements

    size_t off = 0;
    bf16*  hT  = (bf16*)(ws + off);  off += BNC * 2;    // [B,N,C]
    bf16*  QT  = (bf16*)(ws + off);  off += BNC * 2;    // [B,N,C]
    bf16*  Kb  = (bf16*)(ws + off);  off += BNC * 2;    // [B,C,N]
    bf16*  Vb  = (bf16*)(ws + off);  off += BNC * 2;    // [B,C,N]
    float* KV  = (float*)(ws + off); off += BCC * 4;    // [B,C,C] f32
    bf16*  KVb = (bf16*)(ws + off);  off += BCC * 2;    // [B,C,C] bf16
    bf16*  Mb  = (bf16*)(ws + off);  off += BCC * 2;    // [B,C,C] bf16
    bf16*  Wqb = (bf16*)(ws + off);  off += CC * 2;
    bf16*  Wkb = (bf16*)(ws + off);  off += CC * 2;
    bf16*  Wvb = (bf16*)(ws + off);  off += CC * 2;
    bf16*  Wob = (bf16*)(ws + off);  off += CC * 2;

    // weights -> bf16 (n in float4 units: CC/4 = 16384 -> 64 blocks)
    k_f32_to_bf16<<<(int)(CC / 4 / 256), 256, 0, stream>>>(wq, Wqb, (int)(CC / 4));
    k_f32_to_bf16<<<(int)(CC / 4 / 256), 256, 0, stream>>>(wk, Wkb, (int)(CC / 4));
    k_f32_to_bf16<<<(int)(CC / 4 / 256), 256, 0, stream>>>(wv, Wvb, (int)(CC / 4));
    k_f32_to_bf16<<<(int)(CC / 4 / 256), 256, 0, stream>>>(wo, Wob, (int)(CC / 4));

    // GroupNorm -> hT
    k_groupnorm<<<dim3(32, B_DIM), 256, 0, stream>>>(x, gamma, beta, hT);

    // Projections (Q: elu + transpose; K: elu; V: plain)
    k_qkv_gemm<<<dim3(16, 4, B_DIM), 128, 0, stream>>>(Wqb, bq, hT, QT, 1, 1);
    k_qkv_gemm<<<dim3(16, 4, B_DIM), 128, 0, stream>>>(Wkb, bk, hT, Kb, 1, 0);
    k_qkv_gemm<<<dim3(16, 4, B_DIM), 128, 0, stream>>>(Wvb, bv, hT, Vb, 0, 0);

    // kv = K @ V^T  (K-dim = 4096)
    k_kv_gemm<<<dim3(1, 4, B_DIM), 128, 0, stream>>>(Kb, Vb, KV);

    // kv -> bf16
    k_f32_to_bf16<<<(int)(BCC / 4 / 256), 256, 0, stream>>>(KV, KVb, (int)(BCC / 4));

    // M = wo @ kv^T
    k_m_gemm<<<dim3(1, 4, B_DIM), 128, 0, stream>>>(Wob, KVb, Mb);

    // out = x + M @ Q + bo
    k_out_gemm<<<dim3(16, 4, B_DIM), 128, 0, stream>>>(Mb, QT, x, bo, out);
}
